// YunaAudioEncoder_48619029791072
// MI455X (gfx1250) — matrix operand order repaired and checked by reading, not compiled, not run on hardware
//
#include <hip/hip_runtime.h>

typedef __attribute__((ext_vector_type(16))) __bf16 v16bf;
typedef __attribute__((ext_vector_type(8)))  __bf16 v8bf;
typedef __attribute__((ext_vector_type(8)))  float  v8f;

#define DD   1280
#define NCH  12
#define SS   100
#define NHEAD 20
#define HDIM 64
#define FFND 5120
#define TOK  1200

__device__ __forceinline__ v8bf ld8(const __bf16* p){ return *reinterpret_cast<const v8bf*>(p); }
__device__ __forceinline__ void st8(__bf16* p, v8bf v){ *reinterpret_cast<v8bf*>(p) = v; }
__device__ __forceinline__ v8bf zero8(){ v8bf z; for(int i=0;i<8;++i) z[i] = (__bf16)0.0f; return z; }
__device__ __forceinline__ v8f  zero8f(){ v8f z; for(int i=0;i<8;++i) z[i] = 0.0f; return z; }
union U16 { v16bf v; v8bf h[2]; };
__device__ __forceinline__ float gelu_f(float x){ return 0.5f*x*(1.0f+erff(x*0.70710678118654752f)); }

// --- CDNA5 async global->LDS copy (ASYNCcnt-tracked), with safe fallback ----
#if defined(__has_builtin)
#if __has_builtin(__builtin_amdgcn_global_load_async_to_lds_b128)
#define ASYNC_LDS 1
#endif
#endif

#if defined(ASYNC_LDS)
// builtin prototype (from compiler diagnostic): pointee is int vec of 16 bytes
typedef int v4i __attribute__((vector_size(16)));
typedef __attribute__((address_space(1))) v4i* gp4;
typedef __attribute__((address_space(3))) v4i* lp4;
__device__ __forceinline__ void async_cp16(const __bf16* g, __bf16* l){
  // AS3 pointers are 32-bit LDS offsets == low bits of the generic pointer.
  __builtin_amdgcn_global_load_async_to_lds_b128(
      (gp4)(uintptr_t)g, (lp4)(unsigned)(uintptr_t)l, 0, 0);
}
__device__ __forceinline__ void async_wait0(){
#if __has_builtin(__builtin_amdgcn_s_wait_asynccnt)
  __builtin_amdgcn_s_wait_asynccnt(0);
#else
  asm volatile("s_wait_asynccnt 0x0" ::: "memory");
#endif
}
#endif

// ---------------------------------------------------------------------------
// Generic bf16 GEMM: C[M,N] = act(A[M,K] . W[N,K]^T + bias) (+ residual)
// 128x128 block tile, 8 waves x (4x2) tiles of v_wmma_f32_16x16x32_bf16.
// Double-buffered LDS stages, one barrier per 32-wide k-step.
// ---------------------------------------------------------------------------
__global__ __launch_bounds__(256) void gemm_bf16_k(
    const __bf16* __restrict__ A, const __bf16* __restrict__ W,
    const __bf16* __restrict__ bias, const __bf16* __restrict__ res,
    __bf16* __restrict__ C, int M, int N, int K, int act)
{
  __shared__ __bf16 Al[2*128*40];   // 32 k-elems + 8 pad per row, 2 stages
  __shared__ __bf16 Wl[2*128*40];
  const int tid  = threadIdx.x;
  const int lane = tid & 31, wv = tid >> 5;
  const int r = lane & 15, hl = lane >> 4;
  const int mB = blockIdx.x * 128, nB = blockIdx.y * 128;
  const int wm = (wv & 1) * 64, wn = (wv >> 1) * 32;

  v8f acc[4][2];
  for (int i=0;i<4;++i) for (int j=0;j<2;++j) acc[i][j] = zero8f();

  // loader lanes: clamp A row (out-of-range C rows are dropped in epilogue)
  const int lrow = tid >> 1, lhalf = (tid & 1) * 16;
  int gm = mB + lrow; if (gm > M-1) gm = M-1;
  const __bf16* gA = A + (size_t)gm * K + lhalf;
  const __bf16* gW = W + (size_t)(nB + lrow) * K + lhalf;
  __bf16* lA = &Al[lrow*40 + lhalf];
  __bf16* lW = &Wl[lrow*40 + lhalf];
  const int nsteps = K >> 5;

#if defined(ASYNC_LDS)
  // stage k-step s into LDS buffer (s&1) via async DMA (no VGPR staging)
  {
    async_cp16(gA, lA);          async_cp16(gA + 8, lA + 8);
    async_cp16(gW, lW);          async_cp16(gW + 8, lW + 8);
  }
  for (int s = 0; s < nsteps; ++s) {
    async_wait0();
    __syncthreads();
    if (s + 1 < nsteps) {
      int nb = ((s+1) & 1) * 5120;
      const __bf16* a = gA + 32*(s+1); const __bf16* w = gW + 32*(s+1);
      async_cp16(a, lA + nb); async_cp16(a + 8, lA + nb + 8);
      async_cp16(w, lW + nb); async_cp16(w + 8, lW + nb + 8);
    }
    const int buf = (s & 1) * 5120;
#else
  v8bf rA0, rA1, rW0, rW1;
  rA0 = ld8(gA); rA1 = ld8(gA + 8); rW0 = ld8(gW); rW1 = ld8(gW + 8);
  for (int s = 0; s < nsteps; ++s) {
    const int buf = (s & 1) * 5120;
    st8(lA + buf, rA0); st8(lA + buf + 8, rA1);
    st8(lW + buf, rW0); st8(lW + buf + 8, rW1);
    __syncthreads();
    if (s + 1 < nsteps) {
      const __bf16* a = gA + 32*(s+1); const __bf16* w = gW + 32*(s+1);
      rA0 = ld8(a); rA1 = ld8(a + 8); rW0 = ld8(w); rW1 = ld8(w + 8);
    }
#endif
    // B operands (shared across the wave's 4 M tiles)
    U16 b[2];
    for (int j=0;j<2;++j){
      const __bf16* p = &Wl[buf + (wn + 16*j + r)*40 + 8*hl];
      b[j].h[0] = ld8(p); b[j].h[1] = ld8(p + 16);
    }
    for (int i=0;i<4;++i){
      U16 a;  // A striping: chunks at +8*hl and +16+8*hl (doc'd 16-bit A layout)
      const __bf16* p = &Al[buf + (wm + 16*i + r)*40 + 8*hl];
      a.h[0] = ld8(p); a.h[1] = ld8(p + 16);
      for (int j=0;j<2;++j)
        acc[i][j] = __builtin_amdgcn_wmma_f32_16x16x32_bf16(
            false, a.v, false, b[j].v, (short)0, acc[i][j], false, false);
    }
  }

  // epilogue: C layout -> VGPR v, lane l : (row v + 8*hl, col l&15)
  for (int i=0;i<4;++i) for (int j=0;j<2;++j){
    int gcol = nB + wn + 16*j + r;
    float bia = bias ? (float)bias[gcol] : 0.0f;
    for (int v=0; v<8; ++v){
      int grow = mB + wm + 16*i + v + 8*hl;
      if (grow < M) {
        float x = acc[i][j][v] + bia;
        if (act) x = gelu_f(x);
        size_t o = (size_t)grow * N + gcol;
        if (res) x += (float)res[o];
        C[o] = (__bf16)x;
      }
    }
  }
}

// ---------------------------------------------------------------------------
// Fused attention: one block per (chunk, head). 7 waves, 16 q-rows each.
// scores = q.k^T * 1/8 -> softmax(fp32) -> P.V, all via WMMA bf16.
// ---------------------------------------------------------------------------
__global__ __launch_bounds__(224) void attn_k(
    const __bf16* __restrict__ Q, const __bf16* __restrict__ Kb,
    const __bf16* __restrict__ Vb, __bf16* __restrict__ O)
{
  __shared__ __bf16 kl[112*72];     // K rows (padded to 112), 64 dims + pad
  __shared__ __bf16 vt[64*136];     // V transposed: [d][kk], kk padded to 128
  __shared__ __bf16 pl[7*16*128];   // probabilities, row-major per wave strip

  const int tid = threadIdx.x;
  const int c = blockIdx.x / NHEAD, hh = blockIdx.x % NHEAD;
  const size_t base = (size_t)c * SS * DD + (size_t)hh * HDIM;

  for (int idx = tid; idx < 112*64; idx += 224){
    int kk = idx >> 6, d = idx & 63;
    kl[kk*72 + d] = (kk < SS) ? Kb[base + (size_t)kk*DD + d] : (__bf16)0.0f;
  }
  for (int idx = tid; idx < 64*128; idx += 224){
    int d = idx >> 7, kk = idx & 127;
    vt[d*136 + kk] = (kk < SS) ? Vb[base + (size_t)kk*DD + d] : (__bf16)0.0f;
  }
  __syncthreads();

  const int lane = tid & 31, w = tid >> 5;
  const int r = lane & 15, hl = lane >> 4;
  const int qrow = 16*w + r;

  float sc[7][8];
  for (int j=0;j<7;++j){
    v8f acc = zero8f();
    for (int st=0; st<2; ++st){
      U16 a, b;
      if (qrow < SS){
        const __bf16* p = Q + base + (size_t)qrow*DD + 32*st + 8*hl;
        a.h[0] = ld8(p); a.h[1] = ld8(p + 16);
      } else { a.h[0] = zero8(); a.h[1] = zero8(); }
      const __bf16* kp = &kl[(16*j + r)*72 + 32*st + 8*hl];
      b.h[0] = ld8(kp); b.h[1] = ld8(kp + 16);
      acc = __builtin_amdgcn_wmma_f32_16x16x32_bf16(
          false, a.v, false, b.v, (short)0, acc, false, false);
    }
    bool maskc = (16*j + r) >= SS;
    for (int v=0; v<8; ++v) sc[j][v] = maskc ? -1e30f : acc[v]*0.125f;
  }

  // fp32 softmax per row (reduce across 16 lanes of each half-wave)
  for (int v=0; v<8; ++v){
    float m = -1e30f;
    for (int j=0;j<7;++j) m = fmaxf(m, sc[j][v]);
    for (int off=1; off<16; off<<=1) m = fmaxf(m, __shfl_xor(m, off, 32));
    float s = 0.0f;
    for (int j=0;j<7;++j){ float e = __expf(sc[j][v]-m); sc[j][v]=e; s+=e; }
    for (int off=1; off<16; off<<=1) s += __shfl_xor(s, off, 32);
    float inv = 1.0f / s;
    int prow = w*16 + v + 8*hl;
    for (int j=0;j<7;++j) pl[prow*128 + 16*j + r] = (__bf16)(sc[j][v]*inv);
    pl[prow*128 + 112 + r] = (__bf16)0.0f;     // zero the k-pad 112..127
  }
  __syncthreads();

  // O = P . V  (K padded to 128, V^T rows contiguous in LDS)
  for (int t=0;t<4;++t){
    v8f acc = zero8f();
    for (int st=0; st<4; ++st){
      U16 a, b;
      const __bf16* pp = &pl[(w*16 + r)*128 + 32*st + 8*hl];
      a.h[0] = ld8(pp); a.h[1] = ld8(pp + 16);
      const __bf16* vp = &vt[(16*t + r)*136 + 32*st + 8*hl];
      b.h[0] = ld8(vp); b.h[1] = ld8(vp + 16);
      acc = __builtin_amdgcn_wmma_f32_16x16x32_bf16(
          false, a.v, false, b.v, (short)0, acc, false, false);
    }
    for (int v=0; v<8; ++v){
      int row = 16*w + v + 8*hl;
      if (row < SS) O[base + (size_t)row*DD + 16*t + r] = (__bf16)acc[v];
    }
  }
}

// ---------------------------------------------------------------------------
// LayerNorm (one block per row, D=1280)
// ---------------------------------------------------------------------------
__global__ __launch_bounds__(256) void ln_k(const __bf16* __restrict__ X,
    const __bf16* __restrict__ G, const __bf16* __restrict__ B,
    __bf16* __restrict__ Y)
{
  __shared__ float s1[256], s2[256];
  const int row = blockIdx.x, tid = threadIdx.x;
  float xr[5], a = 0.0f, b = 0.0f;
  for (int j=0;j<5;++j){
    float x = (float)X[(size_t)row*DD + tid + 256*j];
    xr[j] = x; a += x; b += x*x;
  }
  s1[tid] = a; s2[tid] = b; __syncthreads();
  for (int st=128; st>0; st>>=1){
    if (tid < st){ s1[tid] += s1[tid+st]; s2[tid] += s2[tid+st]; }
    __syncthreads();
  }
  float mu = s1[0]*(1.0f/DD);
  float var = s2[0]*(1.0f/DD) - mu*mu;
  float rinv = rsqrtf(var + 1e-5f);
  for (int j=0;j<5;++j){
    int e = tid + 256*j;
    Y[(size_t)row*DD + e] = (__bf16)((xr[j]-mu)*rinv*(float)G[e] + (float)B[e]);
  }
}

// AvgPool(2) over time + LayerNorm
__global__ __launch_bounds__(256) void pool_ln_k(const __bf16* __restrict__ Hh,
    const __bf16* __restrict__ G, const __bf16* __restrict__ B,
    __bf16* __restrict__ Y)
{
  __shared__ float s1[256], s2[256];
  const int row = blockIdx.x, tid = threadIdx.x;
  float xr[5], a = 0.0f, b = 0.0f;
  for (int j=0;j<5;++j){
    int e = tid + 256*j;
    float x0 = (float)Hh[(size_t)(2*row)*DD + e];
    float x1 = (float)Hh[(size_t)(2*row+1)*DD + e];
    float x = (float)(__bf16)(0.5f*(x0+x1));   // reference pools in bf16
    xr[j] = x; a += x; b += x*x;
  }
  s1[tid] = a; s2[tid] = b; __syncthreads();
  for (int st=128; st>0; st>>=1){
    if (tid < st){ s1[tid] += s1[tid+st]; s2[tid] += s2[tid+st]; }
    __syncthreads();
  }
  float mu = s1[0]*(1.0f/DD);
  float var = s2[0]*(1.0f/DD) - mu*mu;
  float rinv = rsqrtf(var + 1e-5f);
  for (int j=0;j<5;++j){
    int e = tid + 256*j;
    Y[(size_t)row*DD + e] = (__bf16)((xr[j]-mu)*rinv*(float)G[e] + (float)B[e]);
  }
}

// im2col for conv1: A1[(n*200+t)][ci*3+k] = feat[ci][n*200 + t+k-1] (pad 1/chunk)
__global__ void im2col1_k(const float* __restrict__ F, __bf16* __restrict__ A1)
{
  int idx = blockIdx.x*256 + threadIdx.x;
  if (idx >= 2400*384) return;
  int row = idx / 384, col = idx - row*384;
  int n = row / 200, t = row - n*200;
  int ci = col / 3,  kk = col - ci*3;
  int p = t + kk - 1;
  float v = (p >= 0 && p < 200) ? F[ci*2400 + n*200 + p] : 0.0f;
  A1[idx] = (__bf16)v;
}

// im2col for conv2 (stride 2): A2[(n*100+t)][ci*3+k] = C1[(n*200 + 2t+k-1)][ci]
__global__ void im2col2_k(const __bf16* __restrict__ C1, __bf16* __restrict__ A2)
{
  int idx = blockIdx.x*256 + threadIdx.x;
  if (idx >= 1200*3840) return;
  int row = idx / 3840, col = idx - row*3840;
  int n = row / 100, t = row - n*100;
  int ci = col / 3,  kk = col - ci*3;
  int p = 2*t + kk - 1;
  A2[idx] = (p >= 0 && p < 200) ? C1[(size_t)(n*200 + p)*DD + ci] : (__bf16)0.0f;
}

// add sinusoidal positional embedding (computed on the fly, cast to bf16 first)
__global__ void addpos_k(__bf16* __restrict__ Hh)
{
  int idx = blockIdx.x*256 + threadIdx.x;
  if (idx >= TOK*DD) return;
  int row = idx / DD, d = idx - row*DD;
  int s = row % SS;
  int cc = (d < 640) ? d : d - 640;
  float inv = __expf(-0.0144136784f * (float)cc);   // ln(10000)/639
  float ang = (float)s * inv;
  float p = (d < 640) ? __sinf(ang) : __cosf(ang);
  float pb = (float)(__bf16)p;
  Hh[idx] = (__bf16)((float)Hh[idx] + pb);
}

// ---------------------------------------------------------------------------
extern "C" void kernel_launch(void* const* d_in, const int* in_sizes, int n_in,
                              void* d_out, int out_size, void* d_ws, size_t ws_size,
                              hipStream_t stream)
{
  (void)in_sizes; (void)n_in; (void)out_size; (void)ws_size;
  const float*  feat = (const float*)d_in[0];
  // d_in[1] feature_lens: all multiples of CHUNK -> unused
  const __bf16* c1w = (const __bf16*)d_in[2];
  const __bf16* c1b = (const __bf16*)d_in[3];
  const __bf16* c2w = (const __bf16*)d_in[4];
  const __bf16* c2b = (const __bf16*)d_in[5];
  const __bf16* ln1g = (const __bf16*)d_in[6];
  const __bf16* ln1b = (const __bf16*)d_in[7];
  const __bf16* qw = (const __bf16*)d_in[8];
  const __bf16* qb = (const __bf16*)d_in[9];
  const __bf16* kw = (const __bf16*)d_in[10];
  const __bf16* vw = (const __bf16*)d_in[11];
  const __bf16* vbi = (const __bf16*)d_in[12];
  const __bf16* ow = (const __bf16*)d_in[13];
  const __bf16* ob = (const __bf16*)d_in[14];
  const __bf16* ln2g = (const __bf16*)d_in[15];
  const __bf16* ln2b = (const __bf16*)d_in[16];
  const __bf16* f1w = (const __bf16*)d_in[17];
  const __bf16* f1b = (const __bf16*)d_in[18];
  const __bf16* f2w = (const __bf16*)d_in[19];
  const __bf16* f2b = (const __bf16*)d_in[20];
  const __bf16* lnpg = (const __bf16*)d_in[21];
  const __bf16* lnpb = (const __bf16*)d_in[22];
  const __bf16* pw = (const __bf16*)d_in[23];
  const __bf16* pbias = (const __bf16*)d_in[24];

  char* wsb = (char*)d_ws; size_t off = 0;
  auto alloc = [&](size_t bytes)->void*{
    void* p = wsb + off; off = (off + bytes + 255) & ~(size_t)255; return p;
  };
  __bf16* A1 = (__bf16*)alloc((size_t)2400*384*2);
  __bf16* C1 = (__bf16*)alloc((size_t)2400*1280*2);
  __bf16* A2 = (__bf16*)alloc((size_t)1200*3840*2);
  __bf16* Hh = (__bf16*)alloc((size_t)TOK*DD*2);
  __bf16* Nn = (__bf16*)alloc((size_t)TOK*DD*2);
  __bf16* Qb = (__bf16*)alloc((size_t)TOK*DD*2);
  __bf16* Kc = (__bf16*)alloc((size_t)TOK*DD*2);
  __bf16* Vc = (__bf16*)alloc((size_t)TOK*DD*2);
  __bf16* Ab = (__bf16*)alloc((size_t)TOK*DD*2);
  __bf16* Fb = (__bf16*)alloc((size_t)TOK*FFND*2);
  __bf16* Pb = (__bf16*)alloc((size_t)600*DD*2);

  // conv stem: im2col + WMMA GEMM + exact GELU
  im2col1_k<<<(2400*384+255)/256, 256, 0, stream>>>(feat, A1);
  gemm_bf16_k<<<dim3(19,10), 256, 0, stream>>>(A1, c1w, c1b, nullptr, C1, 2400, 1280, 384, 1);
  im2col2_k<<<(1200*3840+255)/256, 256, 0, stream>>>(C1, A2);
  gemm_bf16_k<<<dim3(10,10), 256, 0, stream>>>(A2, c2w, c2b, nullptr, Hh, TOK, DD, 3840, 1);
  addpos_k<<<(TOK*DD+255)/256, 256, 0, stream>>>(Hh);

  for (int i = 0; i < 32; ++i) {
    const size_t oD = (size_t)i*DD, oDD = (size_t)i*DD*DD, oF = (size_t)i*FFND;
    ln_k<<<TOK, 256, 0, stream>>>(Hh, ln1g + oD, ln1b + oD, Nn);
    gemm_bf16_k<<<dim3(10,10), 256, 0, stream>>>(Nn, qw + oDD, qb + oD, nullptr, Qb, TOK, DD, DD, 0);
    gemm_bf16_k<<<dim3(10,10), 256, 0, stream>>>(Nn, kw + oDD, nullptr, nullptr, Kc, TOK, DD, DD, 0);
    gemm_bf16_k<<<dim3(10,10), 256, 0, stream>>>(Nn, vw + oDD, vbi + oD, nullptr, Vc, TOK, DD, DD, 0);
    attn_k<<<NCH*NHEAD, 224, 0, stream>>>(Qb, Kc, Vc, Ab);
    gemm_bf16_k<<<dim3(10,10), 256, 0, stream>>>(Ab, ow + oDD, ob + oD, Hh, Hh, TOK, DD, DD, 0);
    ln_k<<<TOK, 256, 0, stream>>>(Hh, ln2g + oD, ln2b + oD, Nn);
    gemm_bf16_k<<<dim3(10,40), 256, 0, stream>>>(Nn, f1w + (size_t)i*FFND*DD, f1b + oF, nullptr, Fb, TOK, FFND, DD, 1);
    gemm_bf16_k<<<dim3(10,10), 256, 0, stream>>>(Fb, f2w + (size_t)i*DD*FFND, f2b + oD, Hh, Hh, TOK, DD, FFND, 0);
  }

  pool_ln_k<<<600, 256, 0, stream>>>(Hh, lnpg, lnpb, Pb);
  gemm_bf16_k<<<dim3(5,16), 256, 0, stream>>>(Pb, pw, pbias, nullptr, (__bf16*)d_out, 600, 2048, DD, 0);
}
